// LucyRNNCellTriton_89988154786159
// MI455X (gfx1250) — matrix operand order repaired and checked by reading, not compiled, not run on hardware
//
#include <hip/hip_runtime.h>

// Fused LucyRNN cell for gfx1250 (MI455X).
//
// gates GEMM via v_wmma_f32_16x16x32_f16 (f32 accumulate); the sequential
// scan over T is fused per 16-step chunk so the 192 MB gate tensor never
// touches HBM. HBM traffic ~= x(32MB) + W(6MB) + out(32MB) -> ~3us floor at
// 23.3 TB/s; serial scan chain (2048 dependent steps) is overlapped with the
// next chunk's x prefetch.
//
// Grid: B * (D/16) = 256 blocks. Block: 192 threads = 6 waves (one per gate).
// All 16 A fragments for a chunk are preloaded into registers (32 independent
// ds_load_b128) and a sched_barrier keeps the scheduler from sinking them into
// the WMMA chain -> partial s_wait_dscnt (0x1e..0x0), LDS latency overlapped.

#if defined(__has_builtin)
#  if __has_builtin(__builtin_amdgcn_sched_barrier)
#    define SCHED_FENCE() __builtin_amdgcn_sched_barrier(0)
#  endif
#endif
#ifndef SCHED_FENCE
#  define SCHED_FENCE() asm volatile("" ::: "memory")
#endif

typedef __attribute__((ext_vector_type(16))) _Float16 v16h;
typedef __attribute__((ext_vector_type(8)))  _Float16 v8h;
typedef __attribute__((ext_vector_type(4)))  _Float16 v4h;
typedef __attribute__((ext_vector_type(8)))  float    v8f;
typedef __attribute__((ext_vector_type(4)))  float    v4f;

#define BB      8
#define TT      2048
#define KD      512          // IN_DIM
#define DD      512          // D
#define CHUNK   16           // time steps per WMMA tile (M)
#define NCHUNK  (TT / CHUNK) // 128
#define DTILE   16           // d columns per block (N per gate)
#define NSLICE  (DD / DTILE) // 32
#define NG      6
#define NTHR    192          // 6 wave32
#define XSTR    520          // padded LDS row stride (halves): 1040B, 16B-aligned, bank-skewed
#define GSTR    17           // padded gate-buffer row stride (floats)

__global__ __launch_bounds__(NTHR, 1)
void lucy_rnn_fused(const float* __restrict__ x, const float* __restrict__ h0,
                    const float* __restrict__ s0, const float* __restrict__ W,
                    const float* __restrict__ bias, float* __restrict__ out,
                    float* __restrict__ sT)
{
  // double-buffered x chunk (f16) + gate exchange buffer (f32)
  __shared__ __align__(16) _Float16 xs[2][CHUNK * XSTR];   // 2 * 16640 B
  __shared__ __align__(16) float    gb[NG * CHUNK * GSTR]; // 6528 B

  const int tid  = threadIdx.x;
  const int wv   = tid >> 5;   // wave id == gate id (0..5)
  const int lane = tid & 31;
  const int half = lane >> 4;  // K-half selector in fragments
  const int ln   = lane & 15;  // M row (A) / N col (B,C,D)

  const int bb = blockIdx.x >> 5;                  // batch
  const int d0 = (blockIdx.x & (NSLICE - 1)) * DTILE;

  // Branchless, wave-uniform activation: sigmoid(g)=rcp(1+exp(-g));
  // tanh(g)=2*rcp(1+exp(-2g))-1. Both are scale*rcp(1+exp(a*g))+off.
  const float act_a     = (wv == 2) ? -2.0f : -1.0f;
  const float act_scale = (wv == 2) ?  2.0f :  1.0f;
  const float act_off   = (wv == 2) ? -1.0f :  0.0f;

  // ---- one-time: load this wave's W tile into register-resident B fragments.
  // B column n of the 32x16 fragment = W row (wv*D + d0 + n); lane holds a
  // contiguous 16-K slice per k-block: K = kb*32 + half*16 + e.
  const float* wrow = W + (size_t)(wv * DD + d0 + ln) * KD;
  v16h wfrag[16];
  #pragma unroll
  for (int kb = 0; kb < 16; ++kb) {
    const v4f* wp = (const v4f*)(wrow + kb * 32 + half * 16);
    v4f w0 = wp[0], w1 = wp[1], w2 = wp[2], w3 = wp[3];
    v16h f;
    #pragma unroll
    for (int j = 0; j < 4; ++j) {
      f[j]      = (_Float16)w0[j];
      f[j + 4]  = (_Float16)w1[j];
      f[j + 8]  = (_Float16)w2[j];
      f[j + 12] = (_Float16)w3[j];
    }
    wfrag[kb] = f;
  }
  const float bv = bias[wv * DD + d0 + ln];

  // ---- scan state: carried in registers of wave 0, lanes 0..15
  float h = 0.0f, s = 0.0f;
  if (tid < DTILE) {
    h = h0[bb * DD + d0 + tid];
    s = s0[bb * DD + d0 + tid];
  }

  const float* xb = x + (size_t)bb * TT * KD;

  // ---- preload chunk 0 into xs[0] (fp32 -> f16), all 192 threads
  for (int idx = tid; idx < CHUNK * (KD / 4); idx += NTHR) {
    int r = idx >> 7, c4 = idx & 127;
    v4f v = *(const v4f*)(xb + (size_t)r * KD + c4 * 4);
    v4h hv;
    hv[0] = (_Float16)v[0]; hv[1] = (_Float16)v[1];
    hv[2] = (_Float16)v[2]; hv[3] = (_Float16)v[3];
    *(v4h*)&xs[0][r * XSTR + c4 * 4] = hv;
  }
  __syncthreads();

  for (int c = 0; c < NCHUNK; ++c) {
    const int buf = c & 1;

    // ---- GEMM: acc[16x16] = x_chunk (16x512) * W_tile^T, K in 16 WMMA steps.
    // A fragment (16x32 f16): K=kb*32+half*16+e (contiguous, matches B).
    // Preload ALL fragments (32 independent ds_load_b128), fence the
    // scheduler, then run the WMMA chain with partial dscnt waits.
    const _Float16* xrow = &xs[buf][ln * XSTR + half * 16];
    v8h alo[16], ahi[16];
    #pragma unroll
    for (int kb = 0; kb < 16; ++kb) {
      alo[kb] = *(const v8h*)(xrow + kb * 32);
      ahi[kb] = *(const v8h*)(xrow + kb * 32 + 8);
    }
    SCHED_FENCE();  // keep loads clustered ahead of the WMMA chain

    v8f acc = {0.f, 0.f, 0.f, 0.f, 0.f, 0.f, 0.f, 0.f};
    #pragma unroll
    for (int kb = 0; kb < 16; ++kb) {
      v16h a = __builtin_shufflevector(alo[kb], ahi[kb],
               0, 1, 2, 3, 4, 5, 6, 7, 8, 9, 10, 11, 12, 13, 14, 15);
      acc = __builtin_amdgcn_wmma_f32_16x16x32_f16(
                false, a, false, wfrag[kb], (short)0, acc, false, false);
    }
    SCHED_FENCE();

    // ---- bias + activation in-register (mul,exp,add,rcp,fma per element),
    // then exchange via LDS
    #pragma unroll
    for (int v = 0; v < 8; ++v) {
      float g   = acc[v] + bv;
      float act = act_scale * __builtin_amdgcn_rcpf(1.0f + __expf(act_a * g))
                + act_off;
      gb[wv * (CHUNK * GSTR) + (v + half * 8) * GSTR + ln] = act;  // [g][t][n]
    }
    __syncthreads();

    // ---- overlap: waves 1..5 prefetch next x chunk while wave 0 scans
    if (wv != 0 && c + 1 < NCHUNK) {
      const float* xc = xb + (size_t)(c + 1) * CHUNK * KD;
      for (int idx = tid - 32; idx < CHUNK * (KD / 4); idx += (NTHR - 32)) {
        int r = idx >> 7, c4 = idx & 127;
        v4f v = *(const v4f*)(xc + (size_t)r * KD + c4 * 4);
        v4h hv;
        hv[0] = (_Float16)v[0]; hv[1] = (_Float16)v[1];
        hv[2] = (_Float16)v[2]; hv[3] = (_Float16)v[3];
        *(v4h*)&xs[buf ^ 1][r * XSTR + c4 * 4] = hv;
      }
    }
    if (tid < DTILE) {  // serial scan over the 16 time steps of this chunk
      float* orow = out + (size_t)bb * TT * DD + (size_t)c * CHUNK * DD + d0 + tid;
      #pragma unroll
      for (int t = 0; t < CHUNK; ++t) {
        float ig = gb[0 * (CHUNK * GSTR) + t * GSTR + tid];
        float fg = gb[1 * (CHUNK * GSTR) + t * GSTR + tid];
        float cg = gb[2 * (CHUNK * GSTR) + t * GSTR + tid];
        float og = gb[3 * (CHUNK * GSTR) + t * GSTR + tid];
        float ag = gb[4 * (CHUNK * GSTR) + t * GSTR + tid];
        float wg = gb[5 * (CHUNK * GSTR) + t * GSTR + tid];
        h = fg * h + ig * cg;
        s = ag * s + wg * h;
        orow[(size_t)t * DD] = og * h;
      }
    }
    __syncthreads();
  }

  if (tid < DTILE) sT[bb * DD + d0 + tid] = s;
}

extern "C" void kernel_launch(void* const* d_in, const int* in_sizes, int n_in,
                              void* d_out, int out_size, void* d_ws, size_t ws_size,
                              hipStream_t stream) {
  (void)in_sizes; (void)n_in; (void)out_size; (void)d_ws; (void)ws_size;
  const float* x    = (const float*)d_in[0];
  const float* h0   = (const float*)d_in[1];
  const float* s0   = (const float*)d_in[2];
  const float* W    = (const float*)d_in[3];
  const float* bias = (const float*)d_in[4];
  float* out = (float*)d_out;
  float* sT  = out + (size_t)BB * TT * DD;  // outputs concatenated: out, s_T

  dim3 grid(BB * NSLICE);  // 256
  dim3 block(NTHR);        // 192 = 6 wave32
  lucy_rnn_fused<<<grid, block, 0, stream>>>(x, h0, s0, W, bias, out, sT);
}
